// SampleNet_52390011076907
// MI455X (gfx1250) — matrix operand order repaired and checked by reading, not compiled, not run on hardware
//
#include <hip/hip_runtime.h>
#include <hip/hip_bf16.h>

typedef __attribute__((ext_vector_type(16))) _Float16 v16h;
typedef __attribute__((ext_vector_type(8)))  float    v8f;
typedef __attribute__((ext_vector_type(2)))  float    v2f;

#define D_IN   32
#define H_DIM  64
#define TILE_M 16

#if __has_builtin(__builtin_amdgcn_wmma_f32_16x16x4_f32)
#define HAVE_WMMA_F32X4 1
#else
#define HAVE_WMMA_F32X4 0
#endif

static __device__ __forceinline__ v16h pack4(float4 p0, float4 p1, float4 p2, float4 p3) {
    v16h a;
    a[0]=(_Float16)p0.x;  a[1]=(_Float16)p0.y;  a[2]=(_Float16)p0.z;  a[3]=(_Float16)p0.w;
    a[4]=(_Float16)p1.x;  a[5]=(_Float16)p1.y;  a[6]=(_Float16)p1.z;  a[7]=(_Float16)p1.w;
    a[8]=(_Float16)p2.x;  a[9]=(_Float16)p2.y;  a[10]=(_Float16)p2.z; a[11]=(_Float16)p2.w;
    a[12]=(_Float16)p3.x; a[13]=(_Float16)p3.y; a[14]=(_Float16)p3.z; a[15]=(_Float16)p3.w;
    return a;
}

// ---------------------------------------------------------------------------
// Kernel 1: per-edge MLP scores via WMMA.
// Per wave, 16 edges/iter:
//   h[16x64] = relu( relpos[16x4]*W1pos[4x64] (fp32 WMMA)
//             + Asrc[16x32]*W1a[32x64] + Adst[16x32]*W1b[32x64] (f16 WMMA) + b1 )
//   score    = h @ W2 + b2  (per-lane mult + 16-lane xor reduce)
// ---------------------------------------------------------------------------
__global__ __launch_bounds__(256) void edge_score_kernel(
    const float* __restrict__ x, const float* __restrict__ pos,
    const int* __restrict__ src, const int* __restrict__ dst,
    const float* __restrict__ W1, const float* __restrict__ b1,
    const float* __restrict__ W2, const float* __restrict__ b2,
    float* __restrict__ scores, int E)
{
    // B fragments for W1 rows 0..63: 8 frags (2 K-blocks x 4 N-tiles),
    // per-lane contiguous: frag[f][lane][i], K = kb*32 + 16*(lane>>4) + i,
    // N = t*16 + (lane&15).
    __shared__ __align__(32) _Float16 bfragLds[8 * 32 * 16];

    const int tid = threadIdx.x;
    for (int idx = tid; idx < 8 * 32 * 16; idx += 256) {
        int f    = idx >> 9;
        int rem  = idx & 511;
        int lane = rem >> 4;
        int i    = rem & 15;
        int kb   = f >> 2;
        int t    = f & 3;
        int krow = kb * 32 + ((lane >> 4) << 4) + i;
        int ncol = t * 16 + (lane & 15);
        bfragLds[idx] = (_Float16)W1[krow * H_DIM + ncol];
    }
    __syncthreads();

    const int lane     = tid & 31;
    const int waveId   = tid >> 5;
    const int mrow     = lane & 15;
    const int laneHalf = lane >> 4;

    // Loop-invariant per-lane constants (N = t*16 + mrow).
    float bias[4], w2v[4];
#if HAVE_WMMA_F32X4
    // fp32 B fragment for rel_pos GEMM: B is 4x16 (K x N), K=3 real rows + zero pad.
    // 32-bit B layout: VGPR0 -> K=0 (lanes 0-15) / K=2 (lanes 16-31);
    //                  VGPR1 -> K=1 (lanes 0-15) / K=3 (lanes 16-31).
    v2f bpos[4];
#else
    float w1p0[4], w1p1[4], w1p2[4];
#endif
#pragma unroll
    for (int t = 0; t < 4; ++t) {
        int n   = t * 16 + mrow;
        bias[t] = b1[n];
        w2v[t]  = W2[n];
#if HAVE_WMMA_F32X4
        bpos[t][0] = laneHalf ? W1[66 * H_DIM + n] : W1[64 * H_DIM + n];
        bpos[t][1] = laneHalf ? 0.0f               : W1[65 * H_DIM + n];
#else
        w1p0[t] = W1[64 * H_DIM + n];
        w1p1[t] = W1[65 * H_DIM + n];
        w1p2[t] = W1[66 * H_DIM + n];
#endif
    }
    const float b2s = b2[0];
    const v16h* bfp = (const v16h*)bfragLds;

    const int numTiles = (E + TILE_M - 1) / TILE_M;
    const int gws      = gridDim.x * 8;
    for (int tileIdx = blockIdx.x * 8 + waveId; tileIdx < numTiles; tileIdx += gws) {
        const int e0 = tileIdx * TILE_M;
        const int em = e0 + mrow;
        const bool valid = em < E;
        const int s = valid ? src[em] : 0;
        const int d = valid ? dst[em] : 0;

        // Prefetch next tile's index cachelines (speculative global_prefetch_b8).
        if (em + gws * TILE_M < E) {
            __builtin_prefetch(src + em + gws * TILE_M, 0, 1);
            __builtin_prefetch(dst + em + gws * TILE_M, 0, 1);
        }

        // A fragments (16-bit A 16x32): lane half selects K {0..15} vs {16..31}
        // (contiguous, matching B) -> one contiguous 64B load per row, full row covered
        // exactly once by the two lane halves.
        const float4* xs = (const float4*)(x + (size_t)s * D_IN);
        const float4* xd = (const float4*)(x + (size_t)d * D_IN);
        const int q = laneHalf * 4;
        v16h aS = pack4(xs[q], xs[q + 1], xs[q + 2], xs[q + 3]);
        v16h aD = pack4(xd[q], xd[q + 1], xd[q + 2], xd[q + 3]);

        // rel_pos of this lane's edge (row M = mrow).
        float rx = pos[s * 3 + 0] - pos[d * 3 + 0];
        float ry = pos[s * 3 + 1] - pos[d * 3 + 1];
        float rz = pos[s * 3 + 2] - pos[d * 3 + 2];

#if HAVE_WMMA_F32X4
        // fp32 A fragment 16x4: VGPR0 -> K=0 (half0: rx) / K=2 (half1: rz);
        //                       VGPR1 -> K=1 (half0: ry) / K=3 (half1: 0).
        v2f apos;
        apos[0] = laneHalf ? rz : rx;
        apos[1] = laneHalf ? 0.0f : ry;
#else
        float rxm[8], rym[8], rzm[8];
#pragma unroll
        for (int r = 0; r < 8; ++r) {
            int m = r + 8 * laneHalf;
            rxm[r] = __shfl(rx, m);
            rym[r] = __shfl(ry, m);
            rzm[r] = __shfl(rz, m);
        }
#endif

        float partial[8];
#pragma unroll
        for (int r = 0; r < 8; ++r) partial[r] = 0.0f;

#pragma unroll
        for (int t = 0; t < 4; ++t) {
            v8f c;
#pragma unroll
            for (int r = 0; r < 8; ++r) c[r] = bias[t];

#if HAVE_WMMA_F32X4
            c = __builtin_amdgcn_wmma_f32_16x16x4_f32(false, apos, false, bpos[t],
                                                      (short)0, c, false, false);
#endif
            v16h bS = bfp[t * 32 + lane];            // W1 rows 0..31  (src block)
            c = __builtin_amdgcn_wmma_f32_16x16x32_f16(false, aS, false, bS,
                                                       (short)0, c, false, false);
            v16h bD = bfp[(4 + t) * 32 + lane];      // W1 rows 32..63 (dst block)
            c = __builtin_amdgcn_wmma_f32_16x16x32_f16(false, aD, false, bD,
                                                       (short)0, c, false, false);

#pragma unroll
            for (int r = 0; r < 8; ++r) {
                float v = c[r];
#if !HAVE_WMMA_F32X4
                v += rxm[r] * w1p0[t] + rym[r] * w1p1[t] + rzm[r] * w1p2[t];
#endif
                v = fmaxf(v, 0.0f);                  // ReLU
                partial[r] += v * w2v[t];            // fold W2 per N column
            }
        }

        // Reduce across the 16 N-lanes; lanes 0..15 carry rows 0..7, 16..31 rows 8..15.
#pragma unroll
        for (int r = 0; r < 8; ++r) {
            float p = partial[r];
            p += __shfl_xor(p, 1);
            p += __shfl_xor(p, 2);
            p += __shfl_xor(p, 4);
            p += __shfl_xor(p, 8);
            if (mrow == 0) {
                int e = e0 + r + 8 * laneHalf;
                if (e < E) scores[e] = p + b2s;
            }
        }
    }
}

// ---------------------------------------------------------------------------
// Top-K support kernels (deterministic counting rank, matches lexsort tie-break)
// ---------------------------------------------------------------------------
__global__ void zero_kernel(int* __restrict__ p, int n) {
    for (int i = blockIdx.x * blockDim.x + threadIdx.x; i < n; i += gridDim.x * blockDim.x)
        p[i] = 0;
}

__global__ void hist_kernel(const int* __restrict__ dst, int* __restrict__ counts, int E) {
    for (int e = blockIdx.x * blockDim.x + threadIdx.x; e < E; e += gridDim.x * blockDim.x)
        atomicAdd(&counts[dst[e]], 1);
}

__global__ __launch_bounds__(1024) void scan_kernel(const int* __restrict__ counts,
                                                    int* __restrict__ offsets, int n) {
    __shared__ int sdata[1024];
    __shared__ int carry;
    const int tid = threadIdx.x;
    if (tid == 0) carry = 0;
    __syncthreads();
    for (int base = 0; base < n; base += 1024) {
        int i = base + tid;
        int v = (i < n) ? counts[i] : 0;
        int acc = v;
        sdata[tid] = v;
        __syncthreads();
        for (int off = 1; off < 1024; off <<= 1) {
            int add = (tid >= off) ? sdata[tid - off] : 0;
            __syncthreads();
            acc += add;
            sdata[tid] = acc;
            __syncthreads();
        }
        if (i < n) offsets[i] = carry + acc - v;   // exclusive scan
        int total = sdata[1023];
        __syncthreads();
        if (tid == 0) carry += total;
        __syncthreads();
    }
    if (tid == 0) offsets[n] = carry;
}

__global__ void scatter_kernel(const int* __restrict__ dst, const int* __restrict__ offsets,
                               int* __restrict__ fill, int* __restrict__ list, int E) {
    for (int e = blockIdx.x * blockDim.x + threadIdx.x; e < E; e += gridDim.x * blockDim.x) {
        int d = dst[e];
        int p = offsets[d] + atomicAdd(&fill[d], 1);
        list[p] = e;
    }
}

__global__ void rank_kernel(const int* __restrict__ dst, const float* __restrict__ scores,
                            const int* __restrict__ offsets, const int* __restrict__ list,
                            const int* __restrict__ kp, float* __restrict__ maskOut, int E) {
    const int kk = *kp;
    for (int e = blockIdx.x * blockDim.x + threadIdx.x; e < E; e += gridDim.x * blockDim.x) {
        int d = dst[e];
        float s = scores[e];
        int beg = offsets[d], end = offsets[d + 1];
        int rank = 0;
        for (int j = beg; j < end; ++j) {
            int ep = list[j];
            if (ep == e) continue;
            float sp = scores[ep];
            // stable lexsort((-scores, dst)): higher score first; ties -> lower index first
            rank += (sp > s) || (sp == s && ep < e);
        }
        maskOut[e] = (rank < kk) ? 1.0f : 0.0f;
    }
}

// ---------------------------------------------------------------------------
extern "C" void kernel_launch(void* const* d_in, const int* in_sizes, int n_in,
                              void* d_out, int out_size, void* d_ws, size_t ws_size,
                              hipStream_t stream) {
    const float* x   = (const float*)d_in[0];
    const float* pos = (const float*)d_in[1];
    const int*   ei  = (const int*)d_in[2];
    const int*   kp  = (const int*)d_in[3];
    const float* W1  = (const float*)d_in[4];
    const float* b1  = (const float*)d_in[5];
    const float* W2  = (const float*)d_in[6];
    const float* b2  = (const float*)d_in[7];

    const int N = in_sizes[0] / D_IN;
    const int E = in_sizes[2] / 2;
    const int* src = ei;
    const int* dstp = ei + E;

    float* scores  = (float*)d_out;      // first E outputs: scores
    float* maskOut = scores + E;         // next E outputs: mask (0/1 as float)

    int* counts  = (int*)d_ws;           // [N]
    int* offsets = counts + N;           // [N+1]
    int* fill    = offsets + (N + 1);    // [N]
    int* list    = fill + N;             // [E]

    edge_score_kernel<<<640, 256, 0, stream>>>(x, pos, src, dstp, W1, b1, W2, b2, scores, E);
    zero_kernel<<<256, 256, 0, stream>>>(counts, N);
    zero_kernel<<<256, 256, 0, stream>>>(fill, N);
    hist_kernel<<<2048, 256, 0, stream>>>(dstp, counts, E);
    scan_kernel<<<1, 1024, 0, stream>>>(counts, offsets, N);
    scatter_kernel<<<2048, 256, 0, stream>>>(dstp, offsets, fill, list, E);
    rank_kernel<<<2048, 256, 0, stream>>>(dstp, scores, offsets, list, kp, maskOut, E);
}
